// Hextrema_layer__65859028517619
// MI455X (gfx1250) — matrix operand order repaired and checked by reading, not compiled, not run on hardware
//
#include <hip/hip_runtime.h>
#include <hip/hip_bf16.h>

typedef __attribute__((ext_vector_type(16))) _Float16 v16h;
typedef __attribute__((ext_vector_type(8)))  _Float16 v8h;
typedef __attribute__((ext_vector_type(8)))  float    v8f;

#define BN_EPS 0.001f
#define BATCH  32
#define IMG    (BATCH * 256 * 256)      // 2,097,152 elements
#define FLT_BIG 3.402823466e38f

// ---------------------------------------------------------------------------
// 3x3 min/max filter, SAME with +/-inf padding (zero structuring elements).
// ---------------------------------------------------------------------------
__global__ void morph3_kernel(const float* __restrict__ src, float* __restrict__ dst,
                              int isMax) {
  int idx = blockIdx.x * 256 + threadIdx.x;
  if (idx >= IMG) return;
  int x = idx & 255, y = (idx >> 8) & 255, b = idx >> 16;
  const float* p = src + (b << 16);
  float r = isMax ? -FLT_BIG : FLT_BIG;
  for (int dy = -1; dy <= 1; ++dy) {
    int yy = y + dy; if ((unsigned)yy >= 256u) continue;
    for (int dx = -1; dx <= 1; ++dx) {
      int xx = x + dx; if ((unsigned)xx >= 256u) continue;
      float v = p[(yy << 8) | xx];
      r = isMax ? fmaxf(r, v) : fminf(r, v);
    }
  }
  dst[idx] = r;
}

// ---------------------------------------------------------------------------
// conv1: 3x3 VALID, 1 -> 8 channels, ReLU then BN1. Scalar (K=9 is tiny).
// in: (32,256,256,1)  out: (32,254,254,8) NHWC
// ---------------------------------------------------------------------------
__global__ void conv1_kernel(const float* __restrict__ in, const float* __restrict__ w,
                             const float* __restrict__ bias,
                             const float* __restrict__ g, const float* __restrict__ bb,
                             const float* __restrict__ mn, const float* __restrict__ vr,
                             float* __restrict__ out) {
  const int HO = 254;
  int idx = blockIdx.x * 256 + threadIdx.x;
  int total = BATCH * HO * HO;
  if (idx >= total) return;
  int x = idx % HO; int t = idx / HO; int y = t % HO; int b = t / HO;
  float acc[8];
#pragma unroll
  for (int c = 0; c < 8; ++c) acc[c] = bias[c];
  const float* p = in + (b << 16);
#pragma unroll
  for (int ky = 0; ky < 3; ++ky)
#pragma unroll
    for (int kx = 0; kx < 3; ++kx) {
      float v = p[((y + ky) << 8) | (x + kx)];
#pragma unroll
      for (int c = 0; c < 8; ++c) acc[c] = fmaf(v, w[(ky * 3 + kx) * 8 + c], acc[c]);
    }
#pragma unroll
  for (int c = 0; c < 8; ++c) {
    float r = fmaxf(acc[c], 0.f);
    float sc = g[c] * rsqrtf(vr[c] + BN_EPS);
    out[(long)idx * 8 + c] = (r - mn[c]) * sc + bb[c];
  }
}

// ---------------------------------------------------------------------------
// Pack HWIO f32 weights into the wave32 WMMA B-matrix f16 lane layout.
// B element e of lane: K = chunk*32 + (lane>>4)*8 + (e&7) + (e>>3)*16, N = (lane&15) + t*16.
// wp flat index: ((t*chunks + chunk)*32 + lane)*16 + e
// ---------------------------------------------------------------------------
__global__ void pack_w_kernel(const float* __restrict__ w, _Float16* __restrict__ wp,
                              int CIN, int COUT, int chunks) {
  int NT = COUT >> 4;
  int total = NT * chunks * 32 * 16;
  int i = blockIdx.x * 256 + threadIdx.x;
  if (i >= total) return;
  int e = i & 15; int lane = (i >> 4) & 31; int c = i >> 9;  // c = t*chunks + chunk
  int chunk = c % chunks; int t = c / chunks;
  int K = chunk * 32 + ((lane >> 4) << 3) + (e & 7) + ((e >> 3) << 4);
  int N = (lane & 15) + (t << 4);
  float v = (K < 9 * CIN) ? w[K * COUT + N] : 0.f;
  wp[i] = (_Float16)v;
}

// ---------------------------------------------------------------------------
// Implicit-GEMM 3x3 VALID conv via v_wmma_f32_16x16x32_f16.
// One wave per block (EXEC all ones). M = 16 output pixels in one row,
// N = COUT (NT 16-wide tiles), K = 9*CIN padded to 32 (CH chunks).
// Input patch staged in LDS as f16; every A-fragment 8-half group is a single
// aligned ds_load_b128. Group addresses are computed unconditionally and the
// pad-group case is handled by an integer-offset select into an LDS zero slot
// (tail of the same array) -> pure v_cndmask, no EXEC masking.
// POOL==2 fuses a 2x2 maxpool (pool before BN, as in the reference).
// ---------------------------------------------------------------------------
template <int CIN, int COUT, int POOL>
__global__ void __launch_bounds__(32)
conv3x3_relu_wmma(const float* __restrict__ in, int Hin, int Win,
                  const _Float16* __restrict__ wp,
                  const float* __restrict__ bias,
                  const float* __restrict__ g, const float* __restrict__ bb,
                  const float* __restrict__ mn, const float* __restrict__ vr,
                  int useBN, float* __restrict__ out) {
  constexpr int K9 = 9 * CIN;
  constexpr int CH = (K9 + 31) / 32;
  constexpr int NT = COUT / 16;
  constexpr int PROWS = POOL + 2;              // input rows needed by this block
  constexpr int PATCH = PROWS * 18 * CIN;      // f16 elements staged in LDS
  // lds[0..PATCH-1] = input patch; lds[PATCH..PATCH+7] = zero slot (16B aligned:
  // PATCH is a multiple of 8 since CIN >= 8).
  __shared__ __align__(16) _Float16 lds[PATCH + 8];

  const int lane = threadIdx.x;
  const int Hout = Hin - 2, Wout = Win - 2;
  const int x0 = blockIdx.x * 16;
  const int y0 = blockIdx.y * POOL;
  const int b  = blockIdx.z;
  const int hiHalf = lane >> 4;
  const int mRow = lane & 15;

  // ---- cooperative patch fill (coalesced f32 loads, f16 stores to LDS) ----
  if (lane < 8) lds[PATCH + lane] = (_Float16)0.f;
  for (int flat = lane; flat < PATCH; flat += 32) {
    int row = flat / (18 * CIN);
    int rem = flat - row * (18 * CIN);
    int col = rem / CIN;
    int ci  = rem - col * CIN;
    int xg = x0 + col; if (xg > Win - 1) xg = Win - 1;
    lds[flat] = (_Float16)in[((long)(b * Hin + y0 + row) * Win + xg) * CIN + ci];
  }
  __syncthreads();

  float pool[NT][4];
#pragma unroll
  for (int t = 0; t < NT; ++t)
#pragma unroll
    for (int i = 0; i < 4; ++i) pool[t][i] = -FLT_BIG;

#pragma unroll
  for (int r = 0; r < POOL; ++r) {
    const int y = y0 + r;
    v8f acc[NT];
#pragma unroll
    for (int t = 0; t < NT; ++t) acc[t] = (v8f){0, 0, 0, 0, 0, 0, 0, 0};

#pragma unroll
    for (int chunk = 0; chunk < CH; ++chunk) {
      // A fragment: element e holds K = chunk*32 + hiHalf*8 + (e&7) + (e>>3)*16.
      // Two 8-half groups, each a single ds_load_b128 (zero slot for pad groups).
      v8h grp[2];
#pragma unroll
      for (int gidx = 0; gidx < 2; ++gidx) {
        int K0 = chunk * 32 + gidx * 16 + hiHalf * 8;
        // unconditional address math (pure arithmetic even for pad groups)
        int kyx = K0 / CIN;            // CIN is a power of two -> shift
        int ci  = K0 - kyx * CIN;      // 8-aligned -> 16B-aligned LDS address
        int ky  = kyx / 3;
        int kx  = kyx - 3 * ky;
        int idx = ((ky + r) * 18 + kx + mRow) * CIN + ci;
        int sel = (K0 < K9) ? idx : PATCH;   // v_cndmask, no EXEC masking
        grp[gidx] = *(const v8h*)&lds[sel];
      }
      v16h a = __builtin_shufflevector(grp[0], grp[1],
                                       0, 1, 2, 3, 4, 5, 6, 7,
                                       8, 9, 10, 11, 12, 13, 14, 15);
#pragma unroll
      for (int t = 0; t < NT; ++t) {
        const v16h bmat = *(const v16h*)(wp + ((((long)t * CH + chunk) * 32 + lane) << 4));
        acc[t] = __builtin_amdgcn_wmma_f32_16x16x32_f16(
            false, a, false, bmat, (short)0, acc[t], false, false);
      }
    }

    // D layout: element e8 of lane -> pixel M = e8 + hiHalf*8, channel N = mRow (+t*16)
    if constexpr (POOL == 1) {
#pragma unroll
      for (int t = 0; t < NT; ++t) {
        int n = mRow + t * 16;
        float bi = bias[n];
        float sc = 1.f, sh = 0.f;
        if (useBN) { sc = g[n] * rsqrtf(vr[n] + BN_EPS); sh = bb[n] - mn[n] * sc; }
#pragma unroll
        for (int e8 = 0; e8 < 8; ++e8) {
          int pxo = x0 + e8 + hiHalf * 8;
          if (pxo < Wout) {
            float val = fmaxf(acc[t][e8] + bi, 0.f);
            if (useBN) val = val * sc + sh;
            out[((long)(b * Hout + y) * Wout + pxo) * COUT + n] = val;
          }
        }
      }
    } else {
#pragma unroll
      for (int t = 0; t < NT; ++t) {
        float bi = bias[mRow + t * 16];
#pragma unroll
        for (int e8 = 0; e8 < 8; ++e8) {
          float val = fmaxf(acc[t][e8] + bi, 0.f);
          pool[t][e8 >> 1] = fmaxf(pool[t][e8 >> 1], val);
        }
      }
    }
  }

  if constexpr (POOL == 2) {
    const int Hp = Hout >> 1, Wp = Wout >> 1;
    const int yp = blockIdx.y;
#pragma unroll
    for (int t = 0; t < NT; ++t) {
      int n = mRow + t * 16;
      float sc = 1.f, sh = 0.f;
      if (useBN) { sc = g[n] * rsqrtf(vr[n] + BN_EPS); sh = bb[n] - mn[n] * sc; }
#pragma unroll
      for (int i2 = 0; i2 < 4; ++i2) {
        int xo = (x0 >> 1) + hiHalf * 4 + i2;
        if (xo < Wp) {
          float val = pool[t][i2];
          if (useBN) val = val * sc + sh;
          out[((long)(b * Hp + yp) * Wp + xo) * COUT + n] = val;
        }
      }
    }
  }
}

// ---------------------------------------------------------------------------
// Global max over 61x61 per (batch, channel) then dense 32->1.
// One wave per batch; lane = channel.
// ---------------------------------------------------------------------------
__global__ void __launch_bounds__(32)
gmax_dense_kernel(const float* __restrict__ c4, const float* __restrict__ dw,
                  const float* __restrict__ db, float* __restrict__ hOut,
                  float* __restrict__ hTail) {
  const int P = 61 * 61;
  int b = blockIdx.x, lane = threadIdx.x;
  const float* p = c4 + (long)b * P * 32 + lane;
  float mx = -FLT_BIG;
  for (int i = 0; i < P; ++i) mx = fmaxf(mx, p[(long)i * 32]);
  __shared__ float s[32];
  s[lane] = mx * dw[lane];
  __syncthreads();
  if (lane == 0) {
    float h = db[0];
    for (int j = 0; j < 32; ++j) h += s[j];
    hOut[b] = h;
    hTail[b] = h;
  }
}

// marker = mask - h (h broadcast per batch)
__global__ void marker_init_kernel(const float* __restrict__ mask,
                                   const float* __restrict__ h,
                                   float* __restrict__ m) {
  int idx = blockIdx.x * 256 + threadIdx.x;
  if (idx >= IMG) return;
  m[idx] = mask[idx] - h[idx >> 16];
}

// one reconstruction step: dst = min(dilate3x3(src), mask)
__global__ void hmax_step_kernel(const float* __restrict__ src,
                                 const float* __restrict__ mask,
                                 float* __restrict__ dst) {
  int idx = blockIdx.x * 256 + threadIdx.x;
  if (idx >= IMG) return;
  int x = idx & 255, y = (idx >> 8) & 255, b = idx >> 16;
  const float* p = src + (b << 16);
  float mx = -FLT_BIG;
  for (int dy = -1; dy <= 1; ++dy) {
    int yy = y + dy; if ((unsigned)yy >= 256u) continue;
    for (int dx = -1; dx <= 1; ++dx) {
      int xx = x + dx; if ((unsigned)xx >= 256u) continue;
      mx = fmaxf(mx, p[(yy << 8) | xx]);
    }
  }
  dst[idx] = fminf(mx, mask[idx]);
}

// ---------------------------------------------------------------------------
extern "C" void kernel_launch(void* const* d_in, const int* in_sizes, int n_in,
                              void* d_out, int out_size, void* d_ws, size_t ws_size,
                              hipStream_t stream) {
  const float* x       = (const float*)d_in[0];
  const float* conv1_w = (const float*)d_in[3];
  const float* conv1_b = (const float*)d_in[4];
  const float* conv2_w = (const float*)d_in[5];
  const float* conv2_b = (const float*)d_in[6];
  const float* conv3_w = (const float*)d_in[7];
  const float* conv3_b = (const float*)d_in[8];
  const float* conv4_w = (const float*)d_in[9];
  const float* conv4_b = (const float*)d_in[10];
  const float* bn1_g = (const float*)d_in[11];
  const float* bn1_b = (const float*)d_in[12];
  const float* bn1_m = (const float*)d_in[13];
  const float* bn1_v = (const float*)d_in[14];
  const float* bn2_g = (const float*)d_in[15];
  const float* bn2_b = (const float*)d_in[16];
  const float* bn2_m = (const float*)d_in[17];
  const float* bn2_v = (const float*)d_in[18];
  const float* bn3_g = (const float*)d_in[19];
  const float* bn3_b = (const float*)d_in[20];
  const float* bn3_m = (const float*)d_in[21];
  const float* bn3_v = (const float*)d_in[22];
  const float* dense_w = (const float*)d_in[23];
  const float* dense_b = (const float*)d_in[24];

  float* Dout  = (float*)d_out;
  float* hTail = Dout + IMG;  // h: 32 floats appended after the image output

  // ---- workspace layout (floats) ----
  const long C1N = (long)BATCH * 254 * 254 * 8;    // 16,516,096
  const long C2N = (long)BATCH * 126 * 126 * 16;   //  8,128,512
  float* ws = (float*)d_ws;
  float* A  = ws;            // xc (persists through reconstruction)
  float* T  = A + IMG;       // morph temp / reconstruction ping buffer
  float* C1 = T + IMG;
  float* C2 = C1 + C1N;
  float* C3 = C1;            // reuse: conv1 output dead after conv2
  float* C4 = C2;            // reuse: conv2 output dead after conv3
  float* G  = C2 + C2N;      // h (32 floats)
  _Float16* WP2 = (_Float16*)(G + 32);       // NT=1, CH=3 -> 1536 halves
  _Float16* WP3 = WP2 + 1 * 3 * 32 * 16;     // NT=2, CH=5 -> 5120 halves
  _Float16* WP4 = WP3 + 2 * 5 * 32 * 16;     // NT=2, CH=9 -> 9216 halves

  const int imgBlocks = (IMG + 255) / 256;

  // ---- pack f16 weights into WMMA B layout ----
  pack_w_kernel<<<(1 * 3 * 512 + 255) / 256, 256, 0, stream>>>(conv2_w, WP2, 8, 16, 3);
  pack_w_kernel<<<(2 * 5 * 512 + 255) / 256, 256, 0, stream>>>(conv3_w, WP3, 16, 32, 5);
  pack_w_kernel<<<(2 * 9 * 512 + 255) / 256, 256, 0, stream>>>(conv4_w, WP4, 32, 32, 9);

  // ---- morphological open then close: xc = erode(dilate(dilate(erode(x)))) ----
  morph3_kernel<<<imgBlocks, 256, 0, stream>>>(x, T, 0);     // erode
  morph3_kernel<<<imgBlocks, 256, 0, stream>>>(T, Dout, 1);  // dilate  (xo)
  morph3_kernel<<<imgBlocks, 256, 0, stream>>>(Dout, T, 1);  // dilate
  morph3_kernel<<<imgBlocks, 256, 0, stream>>>(T, A, 0);     // erode   (xc)

  // ---- conv1 + ReLU + BN1 ----
  {
    int total = BATCH * 254 * 254;
    conv1_kernel<<<(total + 255) / 256, 256, 0, stream>>>(
        A, conv1_w, conv1_b, bn1_g, bn1_b, bn1_m, bn1_v, C1);
  }

  // ---- conv2 + ReLU -> maxpool2 -> BN2 (WMMA) ----
  {
    dim3 grid((252 + 15) / 16, 126, BATCH);
    conv3x3_relu_wmma<8, 16, 2><<<grid, 32, 0, stream>>>(
        C1, 254, 254, WP2, conv2_b, bn2_g, bn2_b, bn2_m, bn2_v, 1, C2);
  }

  // ---- conv3 + ReLU + BN3 (WMMA) ----
  {
    dim3 grid((124 + 15) / 16, 124, BATCH);
    conv3x3_relu_wmma<16, 32, 1><<<grid, 32, 0, stream>>>(
        C2, 126, 126, WP3, conv3_b, bn3_g, bn3_b, bn3_m, bn3_v, 1, C3);
  }

  // ---- conv4 + ReLU -> maxpool2 (WMMA, no BN) ----
  {
    dim3 grid((122 + 15) / 16, 61, BATCH);
    conv3x3_relu_wmma<32, 32, 2><<<grid, 32, 0, stream>>>(
        C3, 124, 124, WP4, conv4_b, nullptr, nullptr, nullptr, nullptr, 0, C4);
  }

  // ---- global max + dense -> h (also written to output tail) ----
  gmax_dense_kernel<<<BATCH, 32, 0, stream>>>(C4, dense_w, dense_b, G, hTail);

  // ---- h-maxima reconstruction: 50 ping-pong steps, final lands in d_out ----
  marker_init_kernel<<<imgBlocks, 256, 0, stream>>>(A, G, Dout);
  for (int i = 1; i <= 50; ++i) {
    const float* src = (i & 1) ? Dout : T;
    float* dst       = (i & 1) ? T : Dout;   // i=50 (even) writes Dout
    hmax_step_kernel<<<imgBlocks, 256, 0, stream>>>(src, A, dst);
  }
}
